// TernaryLinear_59562606461169
// MI455X (gfx1250) — compile-verified, matrix operand order checked by reading, not run on hardware
//
#include <hip/hip_runtime.h>
#include <hip/hip_bf16.h>

// ---------------------------------------------------------------------------
// TernaryLinear on gfx1250: out = x @ (tern(W)*scale)^T + bias
//   Path A (enough d_ws): preconvert x->bf16 and W->ternary bf16 once, then a
//     pure bf16 WMMA GEMM. Global->LDS staging uses gfx1250 async-to-LDS
//     copies (ASYNCcnt) when the builtin is available, else b128 reg staging.
//   Path B (tiny d_ws): on-the-fly conversion GEMM (fallback).
// ---------------------------------------------------------------------------

typedef __bf16          v16bf __attribute__((ext_vector_type(16)));
typedef float           v8f   __attribute__((ext_vector_type(8)));
typedef unsigned short  v8u16 __attribute__((ext_vector_type(8)));

#define THRESHOLD 0.05f
#define BM 128
#define BN 128
#define BK 32
#define LDK 40   // padded LDS row stride in bf16 elements (80B: 16B-aligned rows)

#if defined(__gfx1250__) && __has_builtin(__builtin_amdgcn_global_load_async_to_lds_b128)
#define USE_ASYNC 1
#else
#define USE_ASYNC 0
#endif

#if USE_ASYNC
// Builtin parameter pointee type (from the compiler diagnostic): GCC int4.
typedef int gv4i __attribute__((vector_size(4 * sizeof(int))));
typedef __attribute__((address_space(1))) gv4i as1_gv4i;   // global
typedef __attribute__((address_space(3))) gv4i as3_gv4i;   // LDS

// One 16B async copy; IOFFSET (imm) applies to both global and LDS addresses.
// Address spaces are materialized via integer casts: flat-shared low 32 bits
// are the LDS byte offset on amdgcn.
#define ASYNC_B128(GSRC, LDST, OFF)                                           \
  __builtin_amdgcn_global_load_async_to_lds_b128(                             \
      (as1_gv4i*)(unsigned long long)(const void*)(GSRC),                     \
      (as3_gv4i*)(unsigned)(unsigned long long)(void*)(LDST), (OFF), 0)

__device__ __forceinline__ void wait_async0() {
#if __has_builtin(__builtin_amdgcn_s_wait_asynccnt)
  __builtin_amdgcn_s_wait_asynccnt(0);
#else
  asm volatile("s_wait_asynccnt 0x0" ::: "memory");
#endif
}
#endif // USE_ASYNC

__device__ __forceinline__ unsigned short bf16_rne(float f) {
  unsigned int u = __float_as_uint(f);
  u += 0x7fffu + ((u >> 16) & 1u);        // round-to-nearest-even
  return (unsigned short)(u >> 16);
}

__device__ __forceinline__ unsigned short tern_bf16(float w, float thr) {
  return (w >  thr) ? (unsigned short)0x3F80u      // +1.0 bf16
       : (w < -thr) ? (unsigned short)0xBF80u      // -1.0 bf16
       :              (unsigned short)0u;          //  0.0
}

__device__ __forceinline__ v8u16 pack_bf16x8(float4 a, float4 b) {
  v8u16 r;
  r[0]=bf16_rne(a.x); r[1]=bf16_rne(a.y); r[2]=bf16_rne(a.z); r[3]=bf16_rne(a.w);
  r[4]=bf16_rne(b.x); r[5]=bf16_rne(b.y); r[6]=bf16_rne(b.z); r[7]=bf16_rne(b.w);
  return r;
}

__device__ __forceinline__ v8u16 pack_ternx8(float4 a, float4 b, float thr) {
  v8u16 r;
  r[0]=tern_bf16(a.x,thr); r[1]=tern_bf16(a.y,thr); r[2]=tern_bf16(a.z,thr); r[3]=tern_bf16(a.w,thr);
  r[4]=tern_bf16(b.x,thr); r[5]=tern_bf16(b.y,thr); r[6]=tern_bf16(b.z,thr); r[7]=tern_bf16(b.w,thr);
  return r;
}

// -------- kernel: x (f32) -> bf16, 8 elements/thread -----------------------
__global__ void __launch_bounds__(256) cvt_x_kernel(
    const float* __restrict__ X, unsigned short* __restrict__ Xb, long long n) {
  long long i = ((long long)blockIdx.x * 256 + threadIdx.x) * 8;
  if (i + 8 <= n) {
    const float4* g = (const float4*)(X + i);
    ((v8u16*)(Xb + i))[0] = pack_bf16x8(g[0], g[1]);
  }
}

// -------- kernel: per-row scale (+ optional ternary bf16 weight) -----------
__global__ void __launch_bounds__(256) scale_tern_kernel(
    const float* __restrict__ W, float* __restrict__ scale,
    unsigned short* __restrict__ Wt, int K) {
  __shared__ float red[256];
  __shared__ float s_thr;
  const int row = blockIdx.x;
  const int tid = threadIdx.x;
  const float* wr = W + (size_t)row * K;
  float s = 0.f;
  for (int k = tid; k < K; k += 256) s += fabsf(wr[k]);
  red[tid] = s;
  __syncthreads();
  for (int off = 128; off > 0; off >>= 1) {
    if (tid < off) red[tid] += red[tid + off];
    __syncthreads();
  }
  if (tid == 0) {
    float sc = fmaxf(red[0] / (float)K, 1e-8f);
    scale[row] = sc;
    s_thr = THRESHOLD * sc;
  }
  __syncthreads();
  if (Wt != nullptr) {
    const float thr = s_thr;
    unsigned short* tr = Wt + (size_t)row * K;
    for (int c = tid * 8; c + 8 <= K; c += 256 * 8) {
      const float4* g = (const float4*)(wr + c);
      ((v8u16*)(tr + c))[0] = pack_ternx8(g[0], g[1], thr);
    }
  }
}

// -------- tiled WMMA GEMM --------------------------------------------------
// PRE=true : X/W already bf16 (x rounded, W ternarized); async/b128 staging
// PRE=false: X/W are f32; convert/ternarize while staging (fallback)
template <bool PRE>
__global__ void __launch_bounds__(256) tern_gemm_kernel(
    const void* __restrict__ Xv, const void* __restrict__ Wv,
    const float* __restrict__ scale, const float* __restrict__ bias,
    float* __restrict__ Y, int M, int N, int K)
{
  __shared__ __align__(16) unsigned short smA[2][BM * LDK];
  __shared__ __align__(16) unsigned short smB[2][BN * LDK];

  const int tid  = threadIdx.x;
  const int lane = tid & 31;
  const int wid  = tid >> 5;
  const int wm   = wid >> 1;          // 0..3 : wave row within 128x128 tile
  const int wn   = wid & 1;           // 0..1 : wave col
  const int m0   = blockIdx.y * BM;
  const int n0   = blockIdx.x * BN;

  // global->LDS staging: 2 threads per tile row, 16 elements each
  const int lrow = tid >> 1;          // 0..127
  const int lseg = (tid & 1) << 4;    // 0 or 16 elements

  const size_t offA = (size_t)(m0 + lrow) * K + lseg;
  const size_t offB = (size_t)(n0 + lrow) * K + lseg;
  const int    lds_off = lrow * LDK + lseg;   // element offset inside a buffer

  float thr = 0.f;
  if constexpr (!PRE) thr = THRESHOLD * scale[n0 + lrow];

  const v8f vzero = {0.f,0.f,0.f,0.f,0.f,0.f,0.f,0.f};
  v8f acc[2][4];
#pragma unroll
  for (int i = 0; i < 2; ++i)
#pragma unroll
    for (int j = 0; j < 4; ++j) acc[i][j] = vzero;

  // staging registers (unused in the async path)
  v8u16  sa0, sa1, sb0, sb1;
  float4 ra[4], rb[4];

  { // prologue: stage K-step 0 into buffer 0
    if constexpr (PRE) {
#if USE_ASYNC
      const unsigned short* ga = (const unsigned short*)Xv + offA;
      const unsigned short* gb = (const unsigned short*)Wv + offB;
      ASYNC_B128(ga, &smA[0][lds_off], 0);
      ASYNC_B128(ga, &smA[0][lds_off], 16);
      ASYNC_B128(gb, &smB[0][lds_off], 0);
      ASYNC_B128(gb, &smB[0][lds_off], 16);
      wait_async0();
#else
      const v8u16* a8 = (const v8u16*)((const unsigned short*)Xv + offA);
      const v8u16* b8 = (const v8u16*)((const unsigned short*)Wv + offB);
      sa0 = a8[0]; sa1 = a8[1];
      sb0 = b8[0]; sb1 = b8[1];
      v8u16* da = (v8u16*)&smA[0][lds_off];
      da[0] = sa0; da[1] = sa1;
      v8u16* db = (v8u16*)&smB[0][lds_off];
      db[0] = sb0; db[1] = sb1;
#endif
    } else {
      const float4* a4 = (const float4*)((const float*)Xv + offA);
      const float4* b4 = (const float4*)((const float*)Wv + offB);
      ra[0]=a4[0]; ra[1]=a4[1]; ra[2]=a4[2]; ra[3]=a4[3];
      rb[0]=b4[0]; rb[1]=b4[1]; rb[2]=b4[2]; rb[3]=b4[3];
      v8u16* da = (v8u16*)&smA[0][lds_off];
      da[0] = pack_bf16x8(ra[0], ra[1]);
      da[1] = pack_bf16x8(ra[2], ra[3]);
      v8u16* db = (v8u16*)&smB[0][lds_off];
      db[0] = pack_ternx8(rb[0], rb[1], thr);
      db[1] = pack_ternx8(rb[2], rb[3], thr);
    }
  }
  __syncthreads();

  const int ksteps = K / BK;
  const int r  = lane & 15;           // row (A) / col (B) inside 16x16 tile
  const int kg = lane >> 4;           // K-group select per ISA bf16 layout

  for (int ks = 0; ks < ksteps; ++ks) {
    const int  cur  = ks & 1;
    const int  nxt  = cur ^ 1;
    const bool more = (ks + 1) < ksteps;

    if (more) { // stage next K-step; buffer `nxt` was last read before the
                // barrier that ended the previous iteration, so writing is safe
      if constexpr (PRE) {
#if USE_ASYNC
        const unsigned short* ga = (const unsigned short*)Xv + offA + (size_t)(ks+1)*BK;
        const unsigned short* gb = (const unsigned short*)Wv + offB + (size_t)(ks+1)*BK;
        ASYNC_B128(ga, &smA[nxt][lds_off], 0);
        ASYNC_B128(ga, &smA[nxt][lds_off], 16);
        ASYNC_B128(gb, &smB[nxt][lds_off], 0);
        ASYNC_B128(gb, &smB[nxt][lds_off], 16);
#else
        const v8u16* a8 = (const v8u16*)((const unsigned short*)Xv + offA + (size_t)(ks+1)*BK);
        const v8u16* b8 = (const v8u16*)((const unsigned short*)Wv + offB + (size_t)(ks+1)*BK);
        sa0 = a8[0]; sa1 = a8[1];
        sb0 = b8[0]; sb1 = b8[1];
#endif
      } else {
        const float4* a4 = (const float4*)((const float*)Xv + offA + (size_t)(ks+1)*BK);
        const float4* b4 = (const float4*)((const float*)Wv + offB + (size_t)(ks+1)*BK);
        ra[0]=a4[0]; ra[1]=a4[1]; ra[2]=a4[2]; ra[3]=a4[3];
        rb[0]=b4[0]; rb[1]=b4[1]; rb[2]=b4[2]; rb[3]=b4[3];
      }
    }

    // LDS -> fragments, per ISA 16-bit A/B layout: lane holds row/col (lane&15),
    // K halves {kg*8..kg*8+7} and {16+kg*8..16+kg*8+7} -> two b128 loads each.
    const unsigned short* As = smA[cur];
    const unsigned short* Bs = smB[cur];
    v16bf a[2], b[4];
#pragma unroll
    for (int tm = 0; tm < 2; ++tm) {
      const unsigned short* p = As + (wm*32 + tm*16 + r) * LDK + kg*8;
      union { v16bf v; v8u16 h[2]; } u;
      u.h[0] = *(const v8u16*)(p);
      u.h[1] = *(const v8u16*)(p + 16);
      a[tm] = u.v;
    }
#pragma unroll
    for (int tn = 0; tn < 4; ++tn) {
      const unsigned short* p = Bs + (wn*64 + tn*16 + r) * LDK + kg*8;
      union { v16bf v; v8u16 h[2]; } u;
      u.h[0] = *(const v8u16*)(p);
      u.h[1] = *(const v8u16*)(p + 16);
      b[tn] = u.v;
    }

#pragma unroll
    for (int tm = 0; tm < 2; ++tm)
#pragma unroll
      for (int tn = 0; tn < 4; ++tn)
        acc[tm][tn] = __builtin_amdgcn_wmma_f32_16x16x32_bf16(
            false, a[tm], false, b[tn], (short)0, acc[tm][tn], false, false);

    if (more) {
      if constexpr (PRE) {
#if USE_ASYNC
        wait_async0();           // __syncthreads does not cover ASYNCcnt
#else
        v8u16* da = (v8u16*)&smA[nxt][lds_off];
        v8u16* db = (v8u16*)&smB[nxt][lds_off];
        da[0] = sa0; da[1] = sa1;
        db[0] = sb0; db[1] = sb1;
#endif
      } else {
        v8u16* da = (v8u16*)&smA[nxt][lds_off];
        v8u16* db = (v8u16*)&smB[nxt][lds_off];
        da[0] = pack_bf16x8(ra[0], ra[1]);
        da[1] = pack_bf16x8(ra[2], ra[3]);
        db[0] = pack_ternx8(rb[0], rb[1], thr);
        db[1] = pack_ternx8(rb[2], rb[3], thr);
      }
      __syncthreads();
    }
  }

  // epilogue: out = acc * scale[col] + bias[col]
  // C layout: VGPR i, lanes 0-15 -> M=i, lanes 16-31 -> M=8+i; N = lane&15
#pragma unroll
  for (int tn = 0; tn < 4; ++tn) {
    const int   col = n0 + wn*64 + tn*16 + (lane & 15);
    const float s   = scale[col];
    const float bb  = bias[col];
#pragma unroll
    for (int tm = 0; tm < 2; ++tm) {
      const int row0 = m0 + wm*32 + tm*16 + (lane >> 4) * 8;
      float* o = Y + (size_t)row0 * N + col;
#pragma unroll
      for (int i = 0; i < 8; ++i)
        o[(size_t)i * N] = fmaf(acc[tm][tn][i], s, bb);
    }
  }
}

// ---------------------------------------------------------------------------
extern "C" void kernel_launch(void* const* d_in, const int* in_sizes, int n_in,
                              void* d_out, int out_size, void* d_ws, size_t ws_size,
                              hipStream_t stream) {
  const float* x    = (const float*)d_in[0];
  const float* w    = (const float*)d_in[1];
  const float* bias = (const float*)d_in[2];
  float*       out  = (float*)d_out;

  const int N = in_sizes[2];                        // bias length = D_OUT
  const int K = in_sizes[1] / N;                    // weight = N*K
  const int M = in_sizes[0] / K;                    // x = M*K

  const size_t bytesScale = (size_t)N * 4;
  const size_t bytesWt    = (size_t)N * (size_t)K * 2;
  const size_t bytesXb    = (size_t)M * (size_t)K * 2;
  const size_t need       = bytesScale + bytesWt + bytesXb;

  dim3 grid(N / BN, M / BM);
  float* scl = (float*)d_ws;

  if (ws_size >= need) {
    unsigned short* Wt = (unsigned short*)((char*)d_ws + bytesScale);
    unsigned short* Xb = (unsigned short*)((char*)d_ws + bytesScale + bytesWt);
    const long long nX = (long long)M * (long long)K;
    cvt_x_kernel<<<(unsigned)((nX + 2047) / 2048), 256, 0, stream>>>(x, Xb, nX);
    scale_tern_kernel<<<N, 256, 0, stream>>>(w, scl, Wt, K);
    tern_gemm_kernel<true><<<grid, dim3(256), 0, stream>>>(
        (const void*)Xb, (const void*)Wt, scl, bias, out, M, N, K);
  } else {
    scale_tern_kernel<<<N, 256, 0, stream>>>(w, scl, nullptr, K);
    tern_gemm_kernel<false><<<grid, dim3(256), 0, stream>>>(
        (const void*)x, (const void*)w, scl, bias, out, M, N, K);
  }
}
